// AlignmentContrastiveLoss_10806137717449
// MI455X (gfx1250) — matrix operand order repaired and checked by reading, not compiled, not run on hardware
//
#include <hip/hip_runtime.h>
#include <hip/hip_bf16.h>

typedef __attribute__((ext_vector_type(4)))  _Float16 v4h;
typedef __attribute__((ext_vector_type(8)))  _Float16 v8h;
typedef __attribute__((ext_vector_type(16))) _Float16 v16h;
typedef __attribute__((ext_vector_type(8)))  float    v8f;
typedef __attribute__((ext_vector_type(4)))  unsigned int u32x4;
typedef __attribute__((ext_vector_type(8)))  int      i32x8;
typedef __attribute__((ext_vector_type(4)))  int      i32x4;

#define DIMD 1024
#define KCHUNK 512
#define LOSS_MARGIN 0.2f

// Workspace layout:
//   A : [256][48][1024] f16  (im rows 1..36 normalized; invalid & pad rows zeroed)
//   B : [256][32][1024] f16  (s  rows 1..30 normalized; invalid & pad rows zeroed)
//   S : [256][256] f32       (score matrix)

// ---------------------------------------------------------------------------
// Normalize one row (L2 over D=1024), convert to f16, zero invalid/pad rows.
// ---------------------------------------------------------------------------
__global__ __launch_bounds__(256) void norm_rows_kernel(
    const float* __restrict__ src, const int* __restrict__ len,
    _Float16* __restrict__ dst, int srcL, int padL, int validL, int lenOff)
{
  const int row = blockIdx.x;
  const int b = row / padL;
  const int i = row % padL;
  const int tid = threadIdx.x;

  const bool valid = (i < validL) && (i < (len[b] - lenOff));

  float4 v = make_float4(0.f, 0.f, 0.f, 0.f);
  if (valid)
    v = *(const float4*)(src + ((size_t)b * srcL + (i + 1)) * DIMD + tid * 4);

  __shared__ float red[256];
  red[tid] = v.x * v.x + v.y * v.y + v.z * v.z + v.w * v.w;
  __syncthreads();
  for (int s2 = 128; s2 > 0; s2 >>= 1) {
    if (tid < s2) red[tid] += red[tid + s2];
    __syncthreads();
  }
  const float scale = valid ? (1.0f / fmaxf(sqrtf(red[0]), 1e-12f)) : 0.0f;

  v4h o;
  o[0] = (_Float16)(v.x * scale);
  o[1] = (_Float16)(v.y * scale);
  o[2] = (_Float16)(v.z * scale);
  o[3] = (_Float16)(v.w * scale);
  *(v4h*)(dst + ((size_t)b * padL + i) * DIMD + tid * 4) = o;
}

// ---------------------------------------------------------------------------
// Fused GEMM + per-(b,c)-block reduction.
// Workgroup = (pair of b, group of 16 c). Each of 8 waves owns 2 b x 2 c:
// 24 WMMA C-tiles per wave per K-step -> ~427 operand bytes / WMMA.
// A K-chunks (2 b x 48 rows x 512 halfs = 96 KB) staged in LDS by the
// Tensor Data Mover (tensor_load_to_lds + s_wait_tensorcnt).
//   scores[b,c] = sum_{j<30} max_{i<36} (im_b[i,:] . s_c[j,:])
// ---------------------------------------------------------------------------
__global__ __launch_bounds__(256) void gemm_maxsum_kernel(
    const _Float16* __restrict__ A, const _Float16* __restrict__ B,
    float* __restrict__ S)
{
  extern __shared__ _Float16 Alds[];      // [2*48][KCHUNK] halfs = 96 KB

  const int bp   = blockIdx.x;            // b-pair index (0..127)
  const int cg   = blockIdx.y;            // c-group (0..15), 16 c per group
  const int tid  = threadIdx.x;
  const int wave = tid >> 5;
  const int lane = tid & 31;

  const int c0 = cg * 16 + wave * 2;      // this wave's first c
  const _Float16* __restrict__ B0 = B + (size_t)c0 * 32 * DIMD;

  const int half = lane >> 4;             // 0 or 1
  const int r    = lane & 15;

  v8f acc[2][2][3][2] = {};               // [gb][gc][tm][tn]

  const uint64_t gabase =
      (uint64_t)(uintptr_t)(A + (size_t)bp * 2 * 48 * DIMD);
  const uint32_t ldsoff = (uint32_t)(uintptr_t)Alds;

  #pragma unroll 1
  for (int kk = 0; kk < DIMD / KCHUNK; ++kk) {
    // ---- TDM stage: 2D tile, 96 rows x 128 8-byte elems, row stride 256 ----
    if (wave == 0) {
      const uint64_t ga = gabase + (uint64_t)kk * (KCHUNK * 2);  // bytes
      u32x4 g0 = { 1u,                                   // count=1
                   ldsoff,                               // lds_addr
                   (uint32_t)ga,                         // global_addr[31:0]
                   ((uint32_t)(ga >> 32) & 0x01FFFFFFu) | (2u << 30) }; // type=2
      i32x8 g1 = { (int)(3u << 16),                      // data_size = 8B
                   (int)(256u << 16),                    // tensor_dim0 = 256
                   (int)(96u << 16),                     // tensor_dim1 = 96
                   (int)(128u << 16),                    // tile_dim0 = 128
                   (int)96,                              // tile_dim1 = 96
                   (int)256,                             // tensor_dim0_stride
                   0, 0 };
      i32x4 z4 = {0, 0, 0, 0};
#if __clang_major__ >= 23
      i32x8 z8 = {0, 0, 0, 0, 0, 0, 0, 0};
      __builtin_amdgcn_tensor_load_to_lds(g0, g1, z4, z4, z8, 0);
#else
      __builtin_amdgcn_tensor_load_to_lds(g0, g1, z4, z4, 0);
#endif
      __builtin_amdgcn_s_wait_tensorcnt(0);
    }
    __syncthreads();

    for (int k0 = 0; k0 < KCHUNK; k0 += 32) {
      // B fragments (32x16 f16): half 0 -> K 0..15, half 1 -> K 16..31
      v16h bfrag[2][2];
      #pragma unroll
      for (int gc = 0; gc < 2; ++gc)
        #pragma unroll
        for (int tn = 0; tn < 2; ++tn) {
          const _Float16* p = B0 + (size_t)(gc * 32 + tn * 16 + r) * DIMD +
                              kk * KCHUNK + k0 + half * 16;
          v8h lo = *(const v8h*)(p);
          v8h hi = *(const v8h*)(p + 8);
          bfrag[gc][tn] = __builtin_shufflevector(lo, hi,
              0, 1, 2, 3, 4, 5, 6, 7, 8, 9, 10, 11, 12, 13, 14, 15);
        }
      // A fragments (16x32 f16): half 0 -> K 0..15, half 1 -> K 16..31
      #pragma unroll
      for (int gb = 0; gb < 2; ++gb)
        #pragma unroll
        for (int tm = 0; tm < 3; ++tm) {
          const _Float16* p = Alds + (size_t)(gb * 48 + tm * 16 + r) * KCHUNK +
                              k0 + half * 16;
          v8h lo = *(const v8h*)(p);
          v8h hi = *(const v8h*)(p + 8);
          v16h af = __builtin_shufflevector(lo, hi,
              0, 1, 2, 3, 4, 5, 6, 7, 8, 9, 10, 11, 12, 13, 14, 15);
          #pragma unroll
          for (int gc = 0; gc < 2; ++gc)
            #pragma unroll
            for (int tn = 0; tn < 2; ++tn)
              acc[gb][gc][tm][tn] = __builtin_amdgcn_wmma_f32_16x16x32_f16(
                  false, af, false, bfrag[gc][tn],
                  (short)0, acc[gb][gc][tm][tn], false, false);
        }
    }
    __syncthreads();
  }

  // C layout: lane -> N = lane%16 ; VGPR e -> M = tm*16 + e + 8*half.
  // scores[b,c] = sum_{j<30} max_{i<36} C[i,j]
  #pragma unroll
  for (int gb = 0; gb < 2; ++gb)
    #pragma unroll
    for (int gc = 0; gc < 2; ++gc) {
      float contrib = 0.0f;
      #pragma unroll
      for (int tn = 0; tn < 2; ++tn) {
        float cmax = -3.0e38f;
        #pragma unroll
        for (int tm = 0; tm < 3; ++tm) {
          #pragma unroll
          for (int e = 0; e < 8; ++e) {
            const int i = tm * 16 + e + half * 8;
            if (i < 36) cmax = fmaxf(cmax, acc[gb][gc][tm][tn][e]);
          }
        }
        cmax = fmaxf(cmax, __shfl_xor(cmax, 16, 32));   // merge lane halves
        const int j = tn * 16 + r;
        if (half == 0 && j < 30) contrib += cmax;
      }
      #pragma unroll
      for (int off = 16; off > 0; off >>= 1)
        contrib += __shfl_xor(contrib, off, 32);
      if (lane == 0) S[(bp * 2 + gb) * 256 + (c0 + gc)] = contrib;
    }
}

// ---------------------------------------------------------------------------
// Hinge loss with hardest negatives over the 256x256 score matrix.
// ---------------------------------------------------------------------------
__global__ __launch_bounds__(256) void loss_kernel(
    const float* __restrict__ S, float* __restrict__ out)
{
  const int t = threadIdx.x;
  const float db = S[t * 256 + t];
  float rowmax = 0.0f;
  float colmax = 0.0f;
  for (int k = 0; k < 256; ++k) {
    if (k == t) continue;
    const float cs = LOSS_MARGIN + S[t * 256 + k] - db;  // cost_s row t
    rowmax = fmaxf(rowmax, fmaxf(cs, 0.0f));
    const float ci = LOSS_MARGIN + S[k * 256 + t] - db;  // cost_im col t
    colmax = fmaxf(colmax, fmaxf(ci, 0.0f));
  }
  __shared__ float red[256];
  red[t] = rowmax + colmax;
  __syncthreads();
  for (int s2 = 128; s2 > 0; s2 >>= 1) {
    if (t < s2) red[t] += red[t + s2];
    __syncthreads();
  }
  if (t == 0) out[0] = red[0];
}

// ---------------------------------------------------------------------------
extern "C" void kernel_launch(void* const* d_in, const int* in_sizes, int n_in,
                              void* d_out, int out_size, void* d_ws, size_t ws_size,
                              hipStream_t stream) {
  const float* im     = (const float*)d_in[0];
  const float* sseq   = (const float*)d_in[1];
  const int*   im_len = (const int*)d_in[2];
  const int*   s_len  = (const int*)d_in[3];
  float* out = (float*)d_out;

  _Float16* A = (_Float16*)d_ws;                       // 256*48*1024 f16
  _Float16* B = A + (size_t)256 * 48 * DIMD;           // 256*32*1024 f16
  float*    S = (float*)(B + (size_t)256 * 32 * DIMD); // 256*256 f32

  norm_rows_kernel<<<256 * 48, 256, 0, stream>>>(im,   im_len, A, 37, 48, 36, 1);
  norm_rows_kernel<<<256 * 32, 256, 0, stream>>>(sseq, s_len,  B, 33, 32, 30, 3);

  dim3 grid(128, 16);
  gemm_maxsum_kernel<<<grid, 256, 2 * 48 * KCHUNK * (int)sizeof(_Float16), stream>>>(A, B, S);

  loss_kernel<<<1, 256, 0, stream>>>(S, out);
}
